// AttentionHead_20701742367217
// MI455X (gfx1250) — matrix operand order repaired and checked by reading, not compiled, not run on hardware
//
#include <hip/hip_runtime.h>

// AttentionHead: B=8, T=2048, E=1024, D=128, causal softmax, f32 in/out.
//  k0: W -> Wt f16 [3][128][1024] (transposed weights for contiguous A-fragments)
//  k1: Z^T = Wt . x^T  -> Q,K row-major f16 [B,T,D], V transposed f16 [B,D,T]
//      (Q pre-scaled by log2(e)/sqrt(D) so softmax uses exp2)
//  k2: flash attention; K/V chunks staged in LDS by the Tensor Data Mover
//      (tensor_load_to_lds, double-buffered, TENSORcnt-tracked, padded rows
//      for bank-conflict-free ds_load_b128 fragments). S^T = K.Q^T so the
//      probability tile lands directly in B-fragment layout; O^T = V^T.P^T
//      keeps per-query softmax stats lane-local. 16 WMMAs per 32 keys.

#define Bn 8
#define Tn 2048
#define En 1024
#define Dn 128

typedef __attribute__((ext_vector_type(16))) _Float16 v16h;
typedef __attribute__((ext_vector_type(8)))  _Float16 v8h;
typedef __attribute__((ext_vector_type(8)))  float    v8f;
typedef __attribute__((ext_vector_type(4)))  float    v4f;
typedef __attribute__((ext_vector_type(4)))  unsigned int v4u;
typedef __attribute__((ext_vector_type(8)))  int      v8i;
typedef __attribute__((ext_vector_type(4)))  int      v4i;

static __device__ __forceinline__ v16h cat16(v8h a, v8h b) {
  v16h r;
#pragma unroll
  for (int i = 0; i < 8; ++i) { r[i] = a[i]; r[i + 8] = b[i]; }
  return r;
}

static __device__ __forceinline__ v8f wmma_f16(v16h a, v16h b, v8f c) {
  return __builtin_amdgcn_wmma_f32_16x16x32_f16(false, a, false, b,
                                                (short)0, c, false, false);
}

// ---- TDM: issue a 2D f16 tile load global->LDS (ISA ch.8 D# layout) ------
// padIntCode: pad after 2^(code+1) DWORDs; padAmtCode: pad (code+1) DWORDs.
static __device__ __forceinline__ void tdm_load_2d(
    unsigned ldsAddr, const void* gaddr,
    unsigned tensorDim0, unsigned tensorDim1,
    unsigned tileDim0, unsigned tileDim1, unsigned stride0,
    unsigned padIntCode, unsigned padAmtCode) {
  unsigned long long ga = (unsigned long long)(uintptr_t)gaddr;
  v4u g0;
  g0[0] = 1u;                                   // count=1, user descriptor
  g0[1] = ldsAddr;                              // LDS byte address
  g0[2] = (unsigned)ga;                         // global_addr[31:0]
  g0[3] = (unsigned)((ga >> 32) & 0x01FFFFFFu) | 0x80000000u;  // type=2
  v8i g1;
  unsigned w0 = (1u << 16)                      // data_size = 2 bytes
              | (1u << 20)                      // pad_enable
              | (padIntCode << 22) | (padAmtCode << 25);
  g1[0] = (int)w0;
  g1[1] = (int)((tensorDim0 & 0xFFFFu) << 16);
  g1[2] = (int)((tensorDim0 >> 16) | ((tensorDim1 & 0xFFFFu) << 16));
  g1[3] = (int)((tensorDim1 >> 16) | (tileDim0 << 16));
  g1[4] = (int)(tileDim1 & 0xFFFFu);            // tile_dim2 = 0
  g1[5] = (int)stride0;                         // dim0 stride (elements)
  g1[6] = 0;                                    // stride0 hi | stride1 lo
  g1[7] = 0;                                    // stride1 hi
  v4i z4 = {0, 0, 0, 0};
  v8i z8 = {0, 0, 0, 0, 0, 0, 0, 0};
#if __has_builtin(__builtin_amdgcn_tensor_load_to_lds)
  // clang-23 / therock-10.0 form: 6 args (g0, g1, g2, g3, extra, cpol)
  __builtin_amdgcn_tensor_load_to_lds(g0, g1, z4, z4, z8, 0);
#else
  asm volatile("tensor_load_to_lds %0, %1" :: "s"(g0), "s"(g1) : "memory");
#endif
}

static __device__ __forceinline__ void wait_tensorcnt2() {
#if __has_builtin(__builtin_amdgcn_s_wait_tensorcnt)
  __builtin_amdgcn_s_wait_tensorcnt(2);
#else
  asm volatile("s_wait_tensorcnt 0x2" ::: "memory");
#endif
}
static __device__ __forceinline__ void wait_tensorcnt0() {
#if __has_builtin(__builtin_amdgcn_s_wait_tensorcnt)
  __builtin_amdgcn_s_wait_tensorcnt(0);
#else
  asm volatile("s_wait_tensorcnt 0x0" ::: "memory");
#endif
}

// ---------------- kernel 0: weight transpose + f32->f16 -------------------
__global__ void wtrans_kernel(const float* __restrict__ Wq,
                              const float* __restrict__ Wk,
                              const float* __restrict__ Wv,
                              _Float16* __restrict__ Wt) {
  int idx = blockIdx.x * 256 + threadIdx.x;
  if (idx >= 3 * Dn * En) return;
  int m   = idx / (Dn * En);
  int rem = idx - m * (Dn * En);
  int d   = rem / En;
  int e   = rem - d * En;
  const float* W = (m == 0) ? Wq : ((m == 1) ? Wk : Wv);
  Wt[idx] = (_Float16)W[e * Dn + d];             // Wt[m][d][e] = W[e][d]
}

// ---------------- kernel 1: QKV projection (Z^T = Wt . x^T) ---------------
__global__ void __launch_bounds__(256) proj_kernel(
    const float* __restrict__ x, const _Float16* __restrict__ Wt,
    const float* __restrict__ bq, const float* __restrict__ bk,
    const float* __restrict__ bv,
    _Float16* __restrict__ Qo, _Float16* __restrict__ Ko,
    _Float16* __restrict__ Vt) {
  const int wave = threadIdx.x >> 5;
  const int lane = threadIdx.x & 31;
  const int r = lane & 15, h = lane >> 4;
  const int tok = blockIdx.x * 128 + wave * 16 + r;   // flat token = b*T + t
  const float* xrow = x + (size_t)tok * En + 8 * h;
  const float QSCALE = 0.0883883476483184f * 1.44269504088896f;

#pragma unroll
  for (int m = 0; m < 3; ++m) {
    v8f acc[8];
#pragma unroll
    for (int dt = 0; dt < 8; ++dt)
#pragma unroll
      for (int i = 0; i < 8; ++i) acc[dt][i] = 0.0f;

    const _Float16* Wm = Wt + (size_t)m * Dn * En;
    for (int e = 0; e < En; e += 32) {
      const float* xp = xrow + e;
      v4f x0 = *(const v4f*)xp;
      v4f x1 = *(const v4f*)(xp + 4);
      v4f x2 = *(const v4f*)(xp + 16);
      v4f x3 = *(const v4f*)(xp + 20);
      v16h xb;
#pragma unroll
      for (int i = 0; i < 4; ++i) {
        xb[i]      = (_Float16)x0[i];
        xb[4 + i]  = (_Float16)x1[i];
        xb[8 + i]  = (_Float16)x2[i];
        xb[12 + i] = (_Float16)x3[i];
      }
#pragma unroll
      for (int dt = 0; dt < 8; ++dt) {
        const _Float16* wp = Wm + (size_t)(dt * 16 + r) * En + e + 8 * h;
        v16h wa = cat16(*(const v8h*)wp, *(const v8h*)(wp + 16));
        acc[dt] = wmma_f16(wa, xb, acc[dt]);
      }
    }

    const float* bias = (m == 0) ? bq : ((m == 1) ? bk : bv);
    if (m < 2) {
      _Float16* Om = (m == 0) ? Qo : Ko;
      float sc = (m == 0) ? QSCALE : 1.0f;
#pragma unroll
      for (int dt = 0; dt < 8; ++dt) {
        v4f bb0 = *(const v4f*)(bias + dt * 16 + 8 * h);
        v4f bb1 = *(const v4f*)(bias + dt * 16 + 8 * h + 4);
        v8h oh;
#pragma unroll
        for (int i = 0; i < 4; ++i) {
          oh[i]     = (_Float16)((acc[dt][i]     + bb0[i]) * sc);
          oh[4 + i] = (_Float16)((acc[dt][4 + i] + bb1[i]) * sc);
        }
        *(v8h*)(Om + (size_t)tok * Dn + dt * 16 + 8 * h) = oh;
      }
    } else {
      int b = tok >> 11, t = tok & (Tn - 1);
#pragma unroll
      for (int dt = 0; dt < 8; ++dt) {
        v4f bb0 = *(const v4f*)(bias + dt * 16 + 8 * h);
        v4f bb1 = *(const v4f*)(bias + dt * 16 + 8 * h + 4);
#pragma unroll
        for (int i = 0; i < 8; ++i) {
          int d = dt * 16 + 8 * h + i;
          float bv_ = (i < 4) ? bb0[i] : bb1[i - 4];
          Vt[((size_t)(b * Dn + d)) * Tn + t] = (_Float16)(acc[dt][i] + bv_);
        }
      }
    }
  }
}

// ---------------- kernel 2: causal flash attention (TDM-staged) -----------
// LDS layout (per chunk of 64 keys), padded by TDM for conflict-free reads:
//   K buf:  64 rows x (128+8) halfs = 17408 B   (pad 16B per 256B row)
//   V buf: 128 rows x ( 64+8) halfs = 18432 B   (pad 16B per 128B row)
#define KROW 136
#define VROW 72
#define KBUF 17408u
#define VBUF 18432u
#define VOFF (2u * KBUF)

__global__ void __launch_bounds__(256) attn_kernel(
    const _Float16* __restrict__ Q, const _Float16* __restrict__ Kk,
    const _Float16* __restrict__ Vt, float* __restrict__ out) {
  __shared__ __align__(256) char smem[2 * KBUF + 2 * VBUF];

  const int wave = threadIdx.x >> 5, lane = threadIdx.x & 31;
  const int r = lane & 15, h = lane >> 4;
  const int b = blockIdx.y;
  const int qtile = blockIdx.x * 8 + wave;
  const int qbase = qtile * 16;
  const int q = qbase + r;

  // Preload Q^T B-fragments (loop invariant)
  const _Float16* Qp = Q + ((size_t)b * Tn + q) * Dn;
  v16h qf[4];
#pragma unroll
  for (int c = 0; c < 4; ++c) {
    const _Float16* p = Qp + c * 32 + 8 * h;
    qf[c] = cat16(*(const v8h*)p, *(const v8h*)(p + 16));
  }

  const _Float16* Kb = Kk + (size_t)b * Tn * Dn;
  const _Float16* Vb = Vt + (size_t)b * Dn * Tn;
  const unsigned ldsBase = (unsigned)(uintptr_t)&smem[0];

  v8f acc[8];
#pragma unroll
  for (int dt = 0; dt < 8; ++dt)
#pragma unroll
    for (int i = 0; i < 8; ++i) acc[dt][i] = 0.0f;

  float mrun = -3.0e38f, lrun = 0.0f;

  const int nchunks = blockIdx.x * 2 + 2;   // block needs keys < qblock+128

  // Prologue: wave0 issues chunks 0 and 1 via the Tensor Data Mover.
  if (wave == 0) {
#pragma unroll
    for (int c = 0; c < 2; ++c) {
      tdm_load_2d(ldsBase + (c ? KBUF : 0u), Kb + (size_t)c * 64 * Dn,
                  Dn, Tn, Dn, 64, Dn, /*padInt=*/5, /*padAmt=*/3);
      tdm_load_2d(ldsBase + VOFF + (c ? VBUF : 0u), Vb + (size_t)c * 64,
                  Tn, Dn, 64, Dn, Tn, /*padInt=*/4, /*padAmt=*/3);
    }
  }

  for (int c = 0; c < nchunks; ++c) {
    if (wave == 0) {                 // chunk c complete? (2 ops per chunk)
      if (c + 1 < nchunks) wait_tensorcnt2();
      else                 wait_tensorcnt0();
    }
    __syncthreads();                 // chunk c resident in buffer c&1

    if (c * 64 <= qbase + 15) {      // wave-uniform causal skip
      const _Float16* Ks = (const _Float16*)(smem + ((c & 1) ? KBUF : 0u));
      const _Float16* Vs = (const _Float16*)(smem + VOFF + ((c & 1) ? VBUF : 0u));

#pragma unroll
      for (int t = 0; t < 2; ++t) {  // two 32-key substeps
        const int sbase = c * 64 + t * 32;
        if (sbase > qbase + 15) break;

        // S^T tiles from LDS: A = K rows, B = Q^T
        v8f s0, s1;
#pragma unroll
        for (int i = 0; i < 8; ++i) { s0[i] = 0.0f; s1[i] = 0.0f; }
#pragma unroll
        for (int cc = 0; cc < 4; ++cc) {
          const _Float16* k0 = Ks + (t * 32 + r) * KROW + cc * 32 + 8 * h;
          const _Float16* k1 = k0 + 16 * KROW;
          v16h a0 = cat16(*(const v8h*)k0, *(const v8h*)(k0 + 16));
          v16h a1 = cat16(*(const v8h*)k1, *(const v8h*)(k1 + 16));
          s0 = wmma_f16(a0, qf[cc], s0);
          s1 = wmma_f16(a1, qf[cc], s1);
        }

        if (sbase + 31 > qbase) {    // causal mask on the diagonal step
#pragma unroll
          for (int i = 0; i < 8; ++i) {
            int k0i = sbase + 8 * h + i;
            if (k0i > q)      s0[i] = -3.0e38f;
            if (k0i + 16 > q) s1[i] = -3.0e38f;
          }
        }

        // online softmax: lane-local + one xor-16 combine
        float mx = fmaxf(s0[0], s1[0]);
#pragma unroll
        for (int i = 1; i < 8; ++i) mx = fmaxf(mx, fmaxf(s0[i], s1[i]));
        mx = fmaxf(mx, __shfl_xor(mx, 16));
        float mnew = fmaxf(mrun, mx);
        float cf = __builtin_amdgcn_exp2f(mrun - mnew);
        mrun = mnew;

        float rs = 0.0f;
        v8h p0h, p1h;
#pragma unroll
        for (int i = 0; i < 8; ++i) {
          float p0 = __builtin_amdgcn_exp2f(s0[i] - mnew);
          float p1 = __builtin_amdgcn_exp2f(s1[i] - mnew);
          rs += p0 + p1;
          p0h[i] = (_Float16)p0;
          p1h[i] = (_Float16)p1;
        }
        rs += __shfl_xor(rs, 16);
        lrun = lrun * cf + rs;

        // assemble P^T B-fragment (own p0h|p1h; exchanged x0/x1 unused)
        v4u pk0 = __builtin_bit_cast(v4u, p0h);
        v4u pk1 = __builtin_bit_cast(v4u, p1h);
        v4u e0, e1;
#pragma unroll
        for (int j = 0; j < 4; ++j) {
          e0[j] = (unsigned)__shfl_xor((int)pk0[j], 16);
          e1[j] = (unsigned)__shfl_xor((int)pk1[j], 16);
        }
        v8h x0 = __builtin_bit_cast(v8h, e0);
        v8h x1 = __builtin_bit_cast(v8h, e1);
        v8h lo = p0h;
        v8h hi = p1h;
        v16h pb = cat16(lo, hi);

#pragma unroll
        for (int dt = 0; dt < 8; ++dt)
#pragma unroll
          for (int i = 0; i < 8; ++i) acc[dt][i] *= cf;

        // O^T += V^T . P^T, V^T fragments from LDS
#pragma unroll
        for (int dt = 0; dt < 8; ++dt) {
          const _Float16* vp = Vs + (dt * 16 + r) * VROW + t * 32 + 8 * h;
          v16h va = cat16(*(const v8h*)vp, *(const v8h*)(vp + 16));
          acc[dt] = wmma_f16(va, pb, acc[dt]);
        }
      }
    }

    __syncthreads();                 // buffer c&1 free for reuse
    if (wave == 0 && c + 2 < nchunks) {
      int n = c + 2;
      tdm_load_2d(ldsBase + ((n & 1) ? KBUF : 0u), Kb + (size_t)n * 64 * Dn,
                  Dn, Tn, Dn, 64, Dn, 5, 3);
      tdm_load_2d(ldsBase + VOFF + ((n & 1) ? VBUF : 0u), Vb + (size_t)n * 64,
                  Tn, Dn, 64, Dn, Tn, 4, 3);
    }
  }

  float linv = 1.0f / lrun;
  float* op = out + ((size_t)b * Tn + q) * Dn;
#pragma unroll
  for (int dt = 0; dt < 8; ++dt) {
    v4f o0, o1;
#pragma unroll
    for (int i = 0; i < 4; ++i) {
      o0[i] = acc[dt][i] * linv;
      o1[i] = acc[dt][4 + i] * linv;
    }
    *(v4f*)(op + dt * 16 + 8 * h) = o0;
    *(v4f*)(op + dt * 16 + 8 * h + 4) = o1;
  }
}

// ---------------- launch ---------------------------------------------------
extern "C" void kernel_launch(void* const* d_in, const int* in_sizes, int n_in,
                              void* d_out, int out_size, void* d_ws,
                              size_t ws_size, hipStream_t stream) {
  const float* x  = (const float*)d_in[0];
  const float* Wq = (const float*)d_in[1];
  const float* bq = (const float*)d_in[2];
  const float* Wk = (const float*)d_in[3];
  const float* bk = (const float*)d_in[4];
  const float* Wv = (const float*)d_in[5];
  const float* bv = (const float*)d_in[6];
  float* out = (float*)d_out;

  char* ws = (char*)d_ws;
  const size_t WT_BYTES  = (size_t)3 * Dn * En * 2;
  const size_t QKV_BYTES = (size_t)Bn * Tn * Dn * 2;
  _Float16* Wt = (_Float16*)ws;
  _Float16* Qh = (_Float16*)(ws + WT_BYTES);
  _Float16* Kh = (_Float16*)(ws + WT_BYTES + QKV_BYTES);
  _Float16* Vt = (_Float16*)(ws + WT_BYTES + 2 * QKV_BYTES);

  wtrans_kernel<<<(3 * Dn * En + 255) / 256, 256, 0, stream>>>(Wq, Wk, Wv, Wt);
  proj_kernel<<<(Bn * Tn) / 128, 256, 0, stream>>>(x, Wt, bq, bk, bv, Qh, Kh, Vt);
  attn_kernel<<<dim3(Tn / 128, Bn), 256, 0, stream>>>(Qh, Kh, Vt, out);
}
